// EGNN_63333587746884
// MI455X (gfx1250) — matrix operand order repaired and checked by reading, not compiled, not run on hardware
//
#include <hip/hip_runtime.h>
#include <hip/hip_bf16.h>

// ---------------- problem constants (match reference) ----------------
#define NN 50000
#define EE 800000
#define HH 128
#define LL 5
#define EIN 16
#define GG 512
#define BN_EPS 1e-5f

typedef __attribute__((ext_vector_type(16))) _Float16 v16h;
typedef __attribute__((ext_vector_type(8)))  _Float16 v8h;
typedef __attribute__((ext_vector_type(8)))  float    v8f;

__device__ __forceinline__ float atomAddF(float* p, float v) {
    return __hip_atomic_fetch_add(p, v, __ATOMIC_RELAXED, __HIP_MEMORY_SCOPE_AGENT);
}

// ---------------- trivial elementwise kernels ----------------
__global__ __launch_bounds__(256) void k_zero(float* p, long n) {
    long i = (long)blockIdx.x * 256 + threadIdx.x;
    if (i < n) p[i] = 0.f;
}

// Swizzle weights f32 [K,Nc] -> f16 fragments in the contiguous-k WMMA
// B order (k = 16h + s, matching the A loads): fragment f = (nt*(K/32)+kt) is 512 contiguous halfs,
// element index = (f*32 + lane)*16 + hi, source = W[kt*32 + (lane<16?0:16)+hi][nt*16 + (lane&15)]
__global__ __launch_bounds__(256) void k_swzB(const float* __restrict__ W,
                                              _Float16* __restrict__ out,
                                              int K, int Nc) {
    long i = (long)blockIdx.x * 256 + threadIdx.x;
    long total = (long)K * Nc;
    if (i < total) {
        int  hi   = (int)(i & 15);
        int  lane = (int)((i >> 4) & 31);
        long frag = i >> 9;
        int  KT   = K / 32;
        int  kt   = (int)(frag % KT);
        int  nt   = (int)(frag / KT);
        int  k = kt * 32 + ((lane < 16) ? 0 : 16) + hi;
        int  n = nt * 16 + (lane & 15);
        out[i] = (_Float16)W[(long)k * Nc + n];
    }
}

// h[n,c] = node_emb[x[n], c]
__global__ __launch_bounds__(256) void k_embed(const int* __restrict__ x,
                                               const float* __restrict__ emb,
                                               float* __restrict__ h, long n) {
    long i = (long)blockIdx.x * 256 + threadIdx.x;
    if (i < n) h[i] = emb[(long)x[i >> 7] * HH + (i & (HH - 1))];
}

// z = (1+eps[layer]) * h
__global__ __launch_bounds__(256) void k_init_z(const float* __restrict__ h,
                                                const float* __restrict__ eps, int layer,
                                                float* __restrict__ z, long n) {
    long i = (long)blockIdx.x * 256 + threadIdx.x;
    if (i < n) z[i] = (1.f + eps[layer]) * h[i];
}

// ---------------- edge path: e = attr@eW+eb; z[dst] += relu(h[src]+e) ----------------
__global__ __launch_bounds__(256) void k_edge_msg(const int* __restrict__ ei,
                                                  const float* __restrict__ eattr,
                                                  const float* __restrict__ eW,   // [16,128]
                                                  const float* __restrict__ eb,   // [128]
                                                  const float* __restrict__ h,
                                                  float* __restrict__ z) {
    __shared__ float sW[EIN * HH];
    __shared__ float sb[HH];
    int tid = threadIdx.x;
    #pragma unroll
    for (int j = 0; j < (EIN * HH) / 256; ++j) sW[tid + j * 256] = eW[tid + j * 256];
    if (tid < HH) sb[tid] = eb[tid];
    __syncthreads();

    int c    = tid & (HH - 1);
    int pair = tid >> 7;              // 0 or 1
    int e0   = blockIdx.x * 64;
    int eend = e0 + 64; if (eend > EE) eend = EE;
    for (int e = e0 + pair; e < eend; e += 2) {
        int src = ei[e];
        int dst = ei[EE + e];
        const float4* ap4 = (const float4*)(eattr + (long)e * EIN);
        __builtin_prefetch(eattr + (long)(e + 2) * EIN, 0, 1);   // global_prefetch_b8
        float4 a0 = ap4[0], a1 = ap4[1], a2 = ap4[2], a3 = ap4[3];
        float v = sb[c];
        v = fmaf(a0.x, sW[ 0 * HH + c], v); v = fmaf(a0.y, sW[ 1 * HH + c], v);
        v = fmaf(a0.z, sW[ 2 * HH + c], v); v = fmaf(a0.w, sW[ 3 * HH + c], v);
        v = fmaf(a1.x, sW[ 4 * HH + c], v); v = fmaf(a1.y, sW[ 5 * HH + c], v);
        v = fmaf(a1.z, sW[ 6 * HH + c], v); v = fmaf(a1.w, sW[ 7 * HH + c], v);
        v = fmaf(a2.x, sW[ 8 * HH + c], v); v = fmaf(a2.y, sW[ 9 * HH + c], v);
        v = fmaf(a2.z, sW[10 * HH + c], v); v = fmaf(a2.w, sW[11 * HH + c], v);
        v = fmaf(a3.x, sW[12 * HH + c], v); v = fmaf(a3.y, sW[13 * HH + c], v);
        v = fmaf(a3.z, sW[14 * HH + c], v); v = fmaf(a3.w, sW[15 * HH + c], v);
        v += h[(long)src * HH + c];
        v = v > 0.f ? v : 0.f;
        atomAddF(&z[(long)dst * HH + c], v);
    }
}

// ---------------- WMMA GEMM: C[M,Nc] = A[M,K](f32) x Bswz(f16) + bias ----------------
// 256 threads = 8 waves (4 wm x 2 wn). Block tile 128(M) x 64(N).
// Wave tile 32x32: 2 A-fragments x 2 B-fragments -> 4 WMMAs per K-step.
// A staged via LDS (f32->f16), B fragments read straight from swizzled global.
#define A_STRIDE 40   // halfs: 80B rows -> 16B aligned b128 ops, conflict-free strided reads
template<int K, int Nc>
__global__ __launch_bounds__(256)
void k_gemm_wmma(const float* __restrict__ A, const _Float16* __restrict__ Bswz,
                 const float* __restrict__ bias, float* __restrict__ C, int M) {
    constexpr int KT = K / 32;
    __shared__ _Float16 As[128 * A_STRIDE];

    const int tid  = threadIdx.x;
    const int lane = tid & 31;
    const int wave = tid >> 5;
    const int wm   = wave >> 1;          // 0..3 -> 32 rows each
    const int wn   = wave & 1;           // 0..1 -> 32 cols each
    const long blockM = (long)blockIdx.x * 128;
    const int  blockN = blockIdx.y * 64;

    // cooperative A load: thread covers row (tid>>1), halfs [(tid&1)*16, +16)
    const int m_a = tid >> 1;
    const int k_a = (tid & 1) * 16;

    const int m    = lane & 15;
    const int kA   = (lane < 16) ? 0 : 16;  // A-fragment K sub-base
    const int nfr  = lane & 15;
    const int mrow0 = (lane < 16) ? 0 : 8;

    v8f acc00 = {}, acc01 = {}, acc10 = {}, acc11 = {};

    #pragma unroll
    for (int kt = 0; kt < KT; ++kt) {
        // ---- A tile 128x32 f32 -> f16 LDS (clamped rows; no divergence) ----
        {
            long row = blockM + m_a;
            if (row >= M) row = M - 1;       // clamp: garbage rows are masked at store
            const float4* src = (const float4*)(A + row * K + kt * 32 + k_a);
            float4 f0 = src[0], f1 = src[1], f2 = src[2], f3 = src[3];
            v8h h0, h1;
            h0[0] = (_Float16)f0.x; h0[1] = (_Float16)f0.y; h0[2] = (_Float16)f0.z; h0[3] = (_Float16)f0.w;
            h0[4] = (_Float16)f1.x; h0[5] = (_Float16)f1.y; h0[6] = (_Float16)f1.z; h0[7] = (_Float16)f1.w;
            h1[0] = (_Float16)f2.x; h1[1] = (_Float16)f2.y; h1[2] = (_Float16)f2.z; h1[3] = (_Float16)f2.w;
            h1[4] = (_Float16)f3.x; h1[5] = (_Float16)f3.y; h1[6] = (_Float16)f3.z; h1[7] = (_Float16)f3.w;
            *(v8h*)&As[m_a * A_STRIDE + k_a]     = h0;
            *(v8h*)&As[m_a * A_STRIDE + k_a + 8] = h1;
        }
        __syncthreads();

        // ---- B fragments straight from swizzled global (L2-resident) ----
        v16h b0, b1;
        {
            const int nt0 = blockIdx.y * 4 + wn * 2;
            const v8h* bp0 = (const v8h*)(Bswz + ((long)(nt0 * KT + kt) * 32 + lane) * 16);
            const v8h* bp1 = (const v8h*)(Bswz + ((long)((nt0 + 1) * KT + kt) * 32 + lane) * 16);
            v8h l0 = bp0[0], u0 = bp0[1], l1 = bp1[0], u1 = bp1[1];
            b0 = __builtin_shufflevector(l0, u0, 0,1,2,3,4,5,6,7,8,9,10,11,12,13,14,15);
            b1 = __builtin_shufflevector(l1, u1, 0,1,2,3,4,5,6,7,8,9,10,11,12,13,14,15);
        }
        // ---- A fragments from LDS: two b128 loads each ----
        v16h a0, a1;
        {
            int r0 = wm * 32 + m;
            int r1 = r0 + 16;
            v8h l0 = *(const v8h*)&As[r0 * A_STRIDE + kA];
            v8h u0 = *(const v8h*)&As[r0 * A_STRIDE + kA + 8];
            v8h l1 = *(const v8h*)&As[r1 * A_STRIDE + kA];
            v8h u1 = *(const v8h*)&As[r1 * A_STRIDE + kA + 8];
            a0 = __builtin_shufflevector(l0, u0, 0,1,2,3,4,5,6,7,8,9,10,11,12,13,14,15);
            a1 = __builtin_shufflevector(l1, u1, 0,1,2,3,4,5,6,7,8,9,10,11,12,13,14,15);
        }
        acc00 = __builtin_amdgcn_wmma_f32_16x16x32_f16(false, a0, false, b0, (short)0, acc00, false, false);
        acc01 = __builtin_amdgcn_wmma_f32_16x16x32_f16(false, a0, false, b1, (short)0, acc01, false, false);
        acc10 = __builtin_amdgcn_wmma_f32_16x16x32_f16(false, a1, false, b0, (short)0, acc10, false, false);
        acc11 = __builtin_amdgcn_wmma_f32_16x16x32_f16(false, a1, false, b1, (short)0, acc11, false, false);
        __syncthreads();
    }

    // ---- epilogue: D layout -> row (lane<16 ? r : 8+r), col = lane&15 ----
    #pragma unroll
    for (int r = 0; r < 8; ++r) {
        long row0 = blockM + wm * 32 + mrow0 + r;
        long row1 = row0 + 16;
        int  col0 = blockN + wn * 32 + nfr;
        int  col1 = col0 + 16;
        if (row0 < M) {
            C[row0 * Nc + col0] = acc00[r] + bias[col0];
            C[row0 * Nc + col1] = acc01[r] + bias[col1];
        }
        if (row1 < M) {
            C[row1 * Nc + col0] = acc10[r] + bias[col0];
            C[row1 * Nc + col1] = acc11[r] + bias[col1];
        }
    }
}

// ---------------- batchnorm pass 1: per-channel sum / sumsq ----------------
#define BN_ROWS 128
__global__ __launch_bounds__(256) void k_bn_reduce(const float* __restrict__ y, int rows, int C,
                                                   float* __restrict__ sums, float* __restrict__ sumsq) {
    int tid = threadIdx.x;
    int c   = tid % C;
    int ro  = tid / C;
    int rpi = 256 / C;
    float s = 0.f, s2 = 0.f;
    long base = (long)blockIdx.x * BN_ROWS;
    for (int r = ro; r < BN_ROWS; r += rpi) {
        long row = base + r;
        if (row < rows) {
            float v = y[row * C + c];
            s += v; s2 += v * v;
        }
    }
    atomAddF(&sums[c], s);
    atomAddF(&sumsq[c], s2);
}

// ---------------- batchnorm pass 2: normalize + affine + optional relu ----------------
__global__ __launch_bounds__(256) void k_bn_apply(float* __restrict__ y,
                                                  const float* __restrict__ sums,
                                                  const float* __restrict__ sumsq,
                                                  const float* __restrict__ g,
                                                  const float* __restrict__ b,
                                                  long total, int C, float invN, int do_relu) {
    long i = (long)blockIdx.x * 256 + threadIdx.x;
    if (i < total) {
        int c    = (int)(i % C);
        float m  = sums[c] * invN;
        float vv = sumsq[c] * invN - m * m;
        float v  = (y[i] - m) * rsqrtf(vv + BN_EPS) * g[c] + b[c];
        if (do_relu) v = fmaxf(v, 0.f);
        y[i] = v;
    }
}

// ---------------- pooling ----------------
__global__ __launch_bounds__(256) void k_pool_accum(const float* __restrict__ h,
                                                    const int* __restrict__ batch,
                                                    float* __restrict__ pooled,
                                                    float* __restrict__ cnt, long total) {
    long i = (long)blockIdx.x * 256 + threadIdx.x;
    if (i < total) {
        int n = (int)(i >> 7);
        int c = (int)(i & (HH - 1));
        int g = batch[n];
        atomAddF(&pooled[(long)g * HH + c], h[i]);
        if (c == 0) atomAddF(&cnt[g], 1.f);
    }
}

__global__ __launch_bounds__(128) void k_pool_out(const float* __restrict__ pooled,
                                                  const float* __restrict__ cnt,
                                                  const float* __restrict__ out_W,
                                                  const float* __restrict__ out_b,
                                                  float* __restrict__ out) {
    __shared__ float p[HH];
    int g = blockIdx.x, c = threadIdx.x;
    float inv = 1.f / fmaxf(cnt[g], 1.f);
    p[c] = pooled[(long)g * HH + c] * inv;
    __syncthreads();
    float v = out_b[c];
    #pragma unroll 8
    for (int k = 0; k < HH; ++k) v = fmaf(p[k], out_W[k * HH + c], v);
    out[(long)g * HH + c] = v;
}

// ---------------- host orchestration ----------------
static inline unsigned gblk(long n, int b) { return (unsigned)((n + b - 1) / b); }

extern "C" void kernel_launch(void* const* d_in, const int* in_sizes, int n_in,
                              void* d_out, int out_size, void* d_ws, size_t ws_size,
                              hipStream_t stream) {
    const int*   x         = (const int*)  d_in[0];
    const int*   edge_idx  = (const int*)  d_in[1];
    const float* edge_attr = (const float*)d_in[2];
    const int*   batch     = (const int*)  d_in[3];
    const float* node_emb  = (const float*)d_in[4];
    const float* edge_W    = (const float*)d_in[5];
    const float* edge_b    = (const float*)d_in[6];
    const float* eps       = (const float*)d_in[7];
    const float* W1        = (const float*)d_in[8];
    const float* b1        = (const float*)d_in[9];
    const float* g1        = (const float*)d_in[10];
    const float* be1       = (const float*)d_in[11];
    const float* W2        = (const float*)d_in[12];
    const float* b2        = (const float*)d_in[13];
    const float* bn_g      = (const float*)d_in[14];
    const float* bn_b      = (const float*)d_in[15];
    const float* out_W     = (const float*)d_in[16];
    const float* out_b     = (const float*)d_in[17];
    float* out = (float*)d_out;

    // workspace carve-up (256B aligned)
    char* base = (char*)d_ws;
    size_t off = 0;
    auto carve = [&](size_t bytes) { char* p = base + off; off = (off + bytes + 255) & ~(size_t)255; return p; };
    float*    h_buf  = (float*)   carve((size_t)NN * HH * 4);
    float*    z_buf  = (float*)   carve((size_t)NN * HH * 4);
    float*    y1_buf = (float*)   carve((size_t)NN * 2 * HH * 4);
    _Float16* W1s    = (_Float16*)carve((size_t)LL * HH * 2 * HH * 2);  // swizzled
    _Float16* W2s    = (_Float16*)carve((size_t)LL * 2 * HH * HH * 2);  // swizzled
    float*    sums   = (float*)   carve(2 * HH * 4);
    float*    sumsq  = (float*)   carve(2 * HH * 4);
    float*    pooled = (float*)   carve((size_t)GG * HH * 4);
    float*    cnt    = (float*)   carve((size_t)GG * 4);
    (void)ws_size; (void)in_sizes; (void)n_in; (void)out_size;

    const float invN = 1.f / (float)NN;
    const long  perW = (long)HH * 2 * HH;   // elements per layer weight

    // weights -> f16, pre-swizzled into WMMA B-fragment order
    for (int i = 0; i < LL; ++i) {
        k_swzB<<<gblk(perW, 256), 256, 0, stream>>>(W1 + (long)i * perW, W1s + (long)i * perW, HH, 2 * HH);
        k_swzB<<<gblk(perW, 256), 256, 0, stream>>>(W2 + (long)i * perW, W2s + (long)i * perW, 2 * HH, HH);
    }

    // node encoder
    {
        long n = (long)NN * HH;
        k_embed<<<gblk(n, 256), 256, 0, stream>>>(x, node_emb, h_buf, n);
    }

    for (int i = 0; i < LL; ++i) {
        long nNH = (long)NN * HH;
        // z = (1+eps)*h, then scatter-add relu(h[src]+e)
        k_init_z<<<gblk(nNH, 256), 256, 0, stream>>>(h_buf, eps, i, z_buf, nNH);
        k_edge_msg<<<gblk(EE, 64), 256, 0, stream>>>(edge_idx, edge_attr,
                                                     edge_W + (long)i * EIN * HH,
                                                     edge_b + (long)i * HH,
                                                     h_buf, z_buf);
        // y1 = z @ W1 + b1   [N,256]
        {
            dim3 grid(gblk(NN, 128), (2 * HH) / 64);
            k_gemm_wmma<HH, 2 * HH><<<grid, 256, 0, stream>>>(
                z_buf, W1s + (long)i * perW, b1 + (long)i * 2 * HH, y1_buf, NN);
        }
        // BN + relu (inner)
        k_zero<<<gblk(2 * HH, 256), 256, 0, stream>>>(sums, 2 * HH);
        k_zero<<<gblk(2 * HH, 256), 256, 0, stream>>>(sumsq, 2 * HH);
        k_bn_reduce<<<gblk(NN, BN_ROWS), 256, 0, stream>>>(y1_buf, NN, 2 * HH, sums, sumsq);
        k_bn_apply<<<gblk((long)NN * 2 * HH, 256), 256, 0, stream>>>(
            y1_buf, sums, sumsq, g1 + (long)i * 2 * HH, be1 + (long)i * 2 * HH,
            (long)NN * 2 * HH, 2 * HH, invN, 1);
        // h = y1 @ W2 + b2   [N,128]
        {
            dim3 grid(gblk(NN, 128), HH / 64);
            k_gemm_wmma<2 * HH, HH><<<grid, 256, 0, stream>>>(
                y1_buf, W2s + (long)i * perW, b2 + (long)i * HH, h_buf, NN);
        }
        // BN (+ relu unless last layer)
        k_zero<<<gblk(HH, 256), 256, 0, stream>>>(sums, HH);
        k_zero<<<gblk(HH, 256), 256, 0, stream>>>(sumsq, HH);
        k_bn_reduce<<<gblk(NN, BN_ROWS), 256, 0, stream>>>(h_buf, NN, HH, sums, sumsq);
        k_bn_apply<<<gblk(nNH, 256), 256, 0, stream>>>(
            h_buf, sums, sumsq, bn_g + (long)i * HH, bn_b + (long)i * HH,
            nNH, HH, invN, (i < LL - 1) ? 1 : 0);
    }

    // global mean pool + output linear
    k_zero<<<gblk((long)GG * HH, 256), 256, 0, stream>>>(pooled, (long)GG * HH);
    k_zero<<<gblk(GG, 256), 256, 0, stream>>>(cnt, GG);
    k_pool_accum<<<gblk((long)NN * HH, 256), 256, 0, stream>>>(h_buf, batch, pooled, cnt, (long)NN * HH);
    k_pool_out<<<GG, HH, 0, stream>>>(pooled, cnt, out_W, out_b, out);
}
